// GATForSequenceClassification_41875931136076
// MI455X (gfx1250) — matrix operand order repaired and checked by reading, not compiled, not run on hardware
//
#include <hip/hip_runtime.h>
#include <hip/hip_bf16.h>
#include <math.h>

// Problem constants (from reference)
constexpr int B_ = 16, L_ = 256, D_ = 768, H_ = 12, HD_ = 64, NL_ = 4;
constexpr int F_ = 3072, E_ = 50, C_ = 2;

typedef __attribute__((ext_vector_type(16))) __bf16 v16bf;
typedef __attribute__((ext_vector_type(8)))  __bf16 v8bf;
typedef __attribute__((ext_vector_type(8)))  float  v8f;

#if defined(__gfx1250__) && __has_builtin(__builtin_amdgcn_global_load_async_to_lds_b128)
#define ASYNC_LDS 1
#else
#define ASYNC_LDS 0
#endif

#if ASYNC_LDS
// Builtin parameter types (from compiler diagnostic): pointer to int4 in
// global (AS1) / LDS (AS3). reinterpret_cast across address spaces is
// ill-formed in C++, so go through integers: low 32 bits of a generic LDS
// pointer ARE the LDS byte offset.
typedef __attribute__((__vector_size__(16))) int v4i_;
typedef __attribute__((address_space(1))) v4i_* gas1_v4i;
typedef __attribute__((address_space(3))) v4i_* las3_v4i;
static __device__ __forceinline__ void async_cp16(const __bf16* g, __bf16* l) {
  __builtin_amdgcn_global_load_async_to_lds_b128(
      (gas1_v4i)(unsigned long long)(const void*)g,
      (las3_v4i)(unsigned int)(unsigned long long)(const void*)l, 0, 0);
}
#endif

// ---------------------------------------------------------------------------
// LDS-tiled WMMA GEMM (batch=1):  C = A (MxK bf16, lda=K) @ Bt^T (Bt: NxK bf16)
// Block: 256 threads = 8 waves -> 128x64 macro-tile (waves 4x2, 32x32 each).
// K staged through LDS in 64-wide double-buffered tiles; global->LDS uses the
// CDNA5 async-to-LDS path (ASYNCcnt) when available.
// out_mode: 0 = f32 store, 2 = bf16 store with ReLU
// ---------------------------------------------------------------------------
constexpr int MB_ = 128, NB_ = 64, KS_ = 64;
constexpr int LDS_STRIDE = 72;               // padded row stride (elements)
constexpr int ASZ_ = MB_ * LDS_STRIDE;       // 9216 elems
constexpr int BSZ_ = NB_ * LDS_STRIDE;       // 4608 elems

__global__ __launch_bounds__(256) void k_gemm_lds(
    const __bf16* __restrict__ A, const __bf16* __restrict__ Bt,
    const float* __restrict__ bias, void* __restrict__ Cout,
    int K, int ldc, int out_mode)
{
  __shared__ __bf16 lds[2 * (ASZ_ + BSZ_)];
  const int tid  = threadIdx.x;
  const int lane = tid & 31;
  const int wave = tid >> 5;
  const int wm = wave & 3, wn = wave >> 2;   // wave position in 4x2 grid
  const int half = lane >> 4, mn = lane & 15;
  const int m0 = blockIdx.x * MB_;
  const int n0 = blockIdx.y * NB_;
  const int aoff = half * 8;

  // cooperative global->LDS stage of one 64-wide K slice (A:128x64, B:64x64)
  auto stage = [&](int buf, int k0) {
    __bf16* la = lds + buf * (ASZ_ + BSZ_);
    __bf16* lb = la + ASZ_;
#pragma unroll
    for (int i = 0; i < 4; ++i) {                       // A: 1024 b128 chunks
      int c = tid + i * 256;
      int row = c >> 3, col = (c & 7) * 8;
      const __bf16* g = A + (size_t)(m0 + row) * K + k0 + col;
      __bf16* l = la + row * LDS_STRIDE + col;
#if ASYNC_LDS
      async_cp16(g, l);
#else
      *(v8bf*)l = *(const v8bf*)g;
#endif
    }
#pragma unroll
    for (int i = 0; i < 2; ++i) {                       // B: 512 b128 chunks
      int c = tid + i * 256;
      int row = c >> 3, col = (c & 7) * 8;
      const __bf16* g = Bt + (size_t)(n0 + row) * K + k0 + col;
      __bf16* l = lb + row * LDS_STRIDE + col;
#if ASYNC_LDS
      async_cp16(g, l);
#else
      *(v8bf*)l = *(const v8bf*)g;
#endif
    }
  };

  v8f c00 = {}, c01 = {}, c10 = {}, c11 = {};
  stage(0, 0);
#if ASYNC_LDS
  __builtin_amdgcn_s_wait_asynccnt(0);
#endif
  __syncthreads();

  const int nS = K / KS_;
  for (int s = 0; s < nS; ++s) {
    const __bf16* la = lds + (s & 1) * (ASZ_ + BSZ_);
    const __bf16* lb = la + ASZ_;
    if (s + 1 < nS) stage((s + 1) & 1, (s + 1) * KS_);

    const __bf16* As0 = la + (wm * 32 + mn) * LDS_STRIDE;
    const __bf16* As1 = As0 + 16 * LDS_STRIDE;
    const __bf16* Bs0 = lb + (wn * 32 + mn) * LDS_STRIDE + half * 8;
    const __bf16* Bs1 = Bs0 + 16 * LDS_STRIDE;
#pragma unroll
    for (int kk = 0; kk < KS_; kk += 32) {
      v8bf a00 = *(const v8bf*)(As0 + kk + aoff);
      v8bf a01 = *(const v8bf*)(As0 + kk + 16 + aoff);
      v8bf a10 = *(const v8bf*)(As1 + kk + aoff);
      v8bf a11 = *(const v8bf*)(As1 + kk + 16 + aoff);
      v8bf b00 = *(const v8bf*)(Bs0 + kk);
      v8bf b01 = *(const v8bf*)(Bs0 + kk + 16);
      v8bf b10 = *(const v8bf*)(Bs1 + kk);
      v8bf b11 = *(const v8bf*)(Bs1 + kk + 16);
      v16bf a0, a1, b0, b1;
#pragma unroll
      for (int i = 0; i < 8; ++i) {
        a0[i] = a00[i]; a0[i + 8] = a01[i];
        a1[i] = a10[i]; a1[i + 8] = a11[i];
        b0[i] = b00[i]; b0[i + 8] = b01[i];
        b1[i] = b10[i]; b1[i + 8] = b11[i];
      }
      c00 = __builtin_amdgcn_wmma_f32_16x16x32_bf16(false, a0, false, b0, (short)0, c00, false, false);
      c01 = __builtin_amdgcn_wmma_f32_16x16x32_bf16(false, a0, false, b1, (short)0, c01, false, false);
      c10 = __builtin_amdgcn_wmma_f32_16x16x32_bf16(false, a1, false, b0, (short)0, c10, false, false);
      c11 = __builtin_amdgcn_wmma_f32_16x16x32_bf16(false, a1, false, b1, (short)0, c11, false, false);
    }
#if ASYNC_LDS
    __builtin_amdgcn_s_wait_asynccnt(0);
#endif
    __syncthreads();
  }

  const int col0 = n0 + wn * 32 + mn, col1 = col0 + 16;
  const int row0 = m0 + wm * 32 + half * 8, row1 = row0 + 16;
  const float bv0 = bias ? bias[col0] : 0.f;
  const float bv1 = bias ? bias[col1] : 0.f;
  if (out_mode == 0) {
    float* Cb = (float*)Cout;
#pragma unroll
    for (int r = 0; r < 8; ++r) {
      Cb[(size_t)(row0 + r) * ldc + col0] = c00[r] + bv0;
      Cb[(size_t)(row0 + r) * ldc + col1] = c01[r] + bv1;
      Cb[(size_t)(row1 + r) * ldc + col0] = c10[r] + bv0;
      Cb[(size_t)(row1 + r) * ldc + col1] = c11[r] + bv1;
    }
  } else {
    __bf16* Cb = (__bf16*)Cout;
#pragma unroll
    for (int r = 0; r < 8; ++r) {
      float v0 = fmaxf(c00[r] + bv0, 0.f), v1 = fmaxf(c01[r] + bv1, 0.f);
      float v2 = fmaxf(c10[r] + bv0, 0.f), v3 = fmaxf(c11[r] + bv1, 0.f);
      Cb[(size_t)(row0 + r) * ldc + col0] = (__bf16)v0;
      Cb[(size_t)(row0 + r) * ldc + col1] = (__bf16)v1;
      Cb[(size_t)(row1 + r) * ldc + col0] = (__bf16)v2;
      Cb[(size_t)(row1 + r) * ldc + col1] = (__bf16)v3;
    }
  }
}

// ---------------------------------------------------------------------------
// Direct batched WMMA GEMM (attention: small K, operands live in WGP$).
// C[b] = A[b] (MxK) @ Bt[b]^T + bias. Wave = 32x32 macro-tile.
// ---------------------------------------------------------------------------
__global__ __launch_bounds__(256) void k_gemm(
    const __bf16* __restrict__ A, const __bf16* __restrict__ Bt,
    const float* __restrict__ bias, void* __restrict__ Cout,
    int M, int N, int K, int ldc,
    long long sA, long long sB, long long sC, int out_mode)
{
  const int lane = threadIdx.x & 31;
  const int wave = threadIdx.x >> 5;
  const int tilesN = N >> 5;
  const int tilesM = M >> 5;
  int tile = blockIdx.x * 8 + wave;
  if (tile >= tilesM * tilesN) return;           // uniform per-wave exit
  const int tm = tile / tilesN, tn = tile % tilesN;
  const int half = lane >> 4;
  const int mn = lane & 15;

  const __bf16* Ab = A  + (long long)blockIdx.y * sA;
  const __bf16* Bb = Bt + (long long)blockIdx.y * sB;

  const __bf16* arow0 = Ab + (size_t)(tm * 32 + mn) * K;
  const __bf16* arow1 = arow0 + (size_t)16 * K;
  const __bf16* brow0 = Bb + (size_t)(tn * 32 + mn) * K + half * 8;
  const __bf16* brow1 = brow0 + (size_t)16 * K;
  const int aoff = half * 8;

  v8f c00 = {}, c01 = {}, c10 = {}, c11 = {};
  for (int kk = 0; kk < K; kk += 32) {
    v8bf a00 = *(const v8bf*)(arow0 + kk + aoff);
    v8bf a01 = *(const v8bf*)(arow0 + kk + 16 + aoff);
    v8bf a10 = *(const v8bf*)(arow1 + kk + aoff);
    v8bf a11 = *(const v8bf*)(arow1 + kk + 16 + aoff);
    v8bf b00 = *(const v8bf*)(brow0 + kk);
    v8bf b01 = *(const v8bf*)(brow0 + kk + 16);
    v8bf b10 = *(const v8bf*)(brow1 + kk);
    v8bf b11 = *(const v8bf*)(brow1 + kk + 16);
    v16bf a0, a1, b0, b1;
#pragma unroll
    for (int i = 0; i < 8; ++i) {
      a0[i] = a00[i]; a0[i + 8] = a01[i];
      a1[i] = a10[i]; a1[i + 8] = a11[i];
      b0[i] = b00[i]; b0[i + 8] = b01[i];
      b1[i] = b10[i]; b1[i + 8] = b11[i];
    }
    c00 = __builtin_amdgcn_wmma_f32_16x16x32_bf16(false, a0, false, b0, (short)0, c00, false, false);
    c01 = __builtin_amdgcn_wmma_f32_16x16x32_bf16(false, a0, false, b1, (short)0, c01, false, false);
    c10 = __builtin_amdgcn_wmma_f32_16x16x32_bf16(false, a1, false, b0, (short)0, c10, false, false);
    c11 = __builtin_amdgcn_wmma_f32_16x16x32_bf16(false, a1, false, b1, (short)0, c11, false, false);
  }

  const int col0 = tn * 32 + mn, col1 = col0 + 16;
  const int row0 = tm * 32 + half * 8, row1 = row0 + 16;
  const float bv0 = bias ? bias[col0] : 0.f;
  const float bv1 = bias ? bias[col1] : 0.f;
  if (out_mode == 0) {
    float* Cb = (float*)Cout + (long long)blockIdx.y * sC;
#pragma unroll
    for (int r = 0; r < 8; ++r) {
      Cb[(size_t)(row0 + r) * ldc + col0] = c00[r] + bv0;
      Cb[(size_t)(row0 + r) * ldc + col1] = c01[r] + bv1;
      Cb[(size_t)(row1 + r) * ldc + col0] = c10[r] + bv0;
      Cb[(size_t)(row1 + r) * ldc + col1] = c11[r] + bv1;
    }
  } else {
    __bf16* Cb = (__bf16*)Cout + (long long)blockIdx.y * sC;
#pragma unroll
    for (int r = 0; r < 8; ++r) {
      Cb[(size_t)(row0 + r) * ldc + col0] = (__bf16)(c00[r] + bv0);
      Cb[(size_t)(row0 + r) * ldc + col1] = (__bf16)(c01[r] + bv1);
      Cb[(size_t)(row1 + r) * ldc + col0] = (__bf16)(c10[r] + bv0);
      Cb[(size_t)(row1 + r) * ldc + col1] = (__bf16)(c11[r] + bv1);
    }
  }
}

// ---- tiled weight convert+transpose: in f32 [K,N] -> out bf16 [N,K] --------
__global__ __launch_bounds__(256) void k_wcvt(const float* __restrict__ in,
                                              __bf16* __restrict__ out, int K, int N)
{
  __shared__ float sh[32][33];
  const int tilesN = N / 32;
  const int tk = blockIdx.x / tilesN, tn = blockIdx.x % tilesN;
  const int tx = threadIdx.x & 31, ty = threadIdx.x >> 5;   // 32 x 8
#pragma unroll
  for (int r = 0; r < 32; r += 8)
    sh[ty + r][tx] = in[(size_t)(tk * 32 + ty + r) * N + tn * 32 + tx];
  __syncthreads();
#pragma unroll
  for (int r = 0; r < 32; r += 8)
    out[(size_t)(tn * 32 + ty + r) * K + tk * 32 + tx] = (__bf16)sh[tx][ty + r];
}

// ---- edge table zero-padded to [64,64] bf16 --------------------------------
__global__ __launch_bounds__(256) void k_edgepad(const float* __restrict__ ee,
                                                 __bf16* __restrict__ out)
{
  int idx = blockIdx.x * 256 + threadIdx.x;
  if (idx >= 64 * 64) return;
  int e = idx / 64;
  out[idx] = (__bf16)((e < E_) ? ee[idx] : 0.f);  // ee is [E,HD] with HD==64
}

// ---- embedding gather + sinusoidal positional encoding ---------------------
__global__ __launch_bounds__(256) void k_embed(const int* __restrict__ ids,
    const float* __restrict__ wemb, float* __restrict__ h, __bf16* __restrict__ hb)
{
  int row = blockIdx.x;                 // b*L + l
  int l = row % L_;
  int id = ids[row];
#pragma unroll
  for (int t = 0; t < 3; ++t) {
    int d = threadIdx.x + t * 256;
    int dp = d & ~1;
    float div = __expf(-logf(10000.0f) * (float)dp / (float)D_);
    float ang = (float)l * div;
    float pe = (d & 1) ? cosf(ang) : sinf(ang);
    float v = wemb[(size_t)id * D_ + d] + pe;
    h[(size_t)row * D_ + d] = v;
    hb[(size_t)row * D_ + d] = (__bf16)v;
  }
}

// ---- repack [B*L, D] f32 -> [B,H,L,HD] bf16 --------------------------------
__global__ __launch_bounds__(256) void k_pack_heads(const float* __restrict__ in,
                                                    __bf16* __restrict__ out)
{
  size_t idx = (size_t)blockIdx.x * 256 + threadIdx.x;
  int col = (int)(idx % D_);
  size_t row = idx / D_;
  int h = col / HD_, d = col % HD_;
  int b = (int)(row / L_), l = (int)(row % L_);
  out[(((size_t)(b * H_ + h)) * L_ + l) * HD_ + d] = (__bf16)in[idx];
}

// ---- repack [B*L, D] f32 -> [B,H,HD,L] bf16 (v transposed) -----------------
__global__ __launch_bounds__(256) void k_pack_vT(const float* __restrict__ in,
                                                 __bf16* __restrict__ out)
{
  size_t idx = (size_t)blockIdx.x * 256 + threadIdx.x;
  int col = (int)(idx % D_);
  size_t row = idx / D_;
  int h = col / HD_, d = col % HD_;
  int b = (int)(row / L_), l = (int)(row % L_);
  out[(((size_t)(b * H_ + h)) * HD_ + d) * L_ + l] = (__bf16)in[idx];
}

// ---- repack [B,H,L,HD] f32 -> [B*L, D] bf16 --------------------------------
__global__ __launch_bounds__(256) void k_pack_merge(const float* __restrict__ in,
                                                    __bf16* __restrict__ out)
{
  size_t idx = (size_t)blockIdx.x * 256 + threadIdx.x;
  int d = (int)(idx % HD_);
  size_t r = idx / HD_;
  int l = (int)(r % L_);
  size_t r2 = r / L_;
  int h = (int)(r2 % H_);
  int b = (int)(r2 / H_);
  out[((size_t)(b * L_ + l)) * D_ + h * HD_ + d] = (__bf16)in[idx];
}

// ---- fused edge-gather + mask + scale + softmax -> bf16 attention ----------
__global__ __launch_bounds__(256) void k_softmax(const float* __restrict__ scores,
    const float* __restrict__ p, const int* __restrict__ et,
    const unsigned char* __restrict__ adj, __bf16* __restrict__ att)
{
  __shared__ float red[256];
  int idx = blockIdx.x;                 // (b*H + h)*L + i
  int i = idx % L_;
  int bh = idx / L_;
  int b = bh / H_;
  int j = threadIdx.x;
  size_t bij = ((size_t)b * L_ + i) * L_ + j;
  float s = scores[(size_t)idx * L_ + j];
  s += p[((size_t)bh * L_ + i) * 64 + et[bij]];     // edge term via gather
  float val = adj[bij] ? s * 0.125f : -1e9f;        // 1/sqrt(64) = 0.125
  red[j] = val; __syncthreads();
  for (int o = 128; o > 0; o >>= 1) { if (j < o) red[j] = fmaxf(red[j], red[j + o]); __syncthreads(); }
  float mx = red[0]; __syncthreads();
  float e = __expf(val - mx);
  red[j] = e; __syncthreads();
  for (int o = 128; o > 0; o >>= 1) { if (j < o) red[j] += red[j + o]; __syncthreads(); }
  att[(size_t)idx * L_ + j] = (__bf16)(e / red[0]);
}

// ---- fused residual-add + LayerNorm -> h (f32) and hb (bf16) ---------------
__global__ __launch_bounds__(256) void k_ln(const float* __restrict__ delta,
    float* __restrict__ h, __bf16* __restrict__ hb,
    const float* __restrict__ g, const float* __restrict__ bvec)
{
  __shared__ float red[256];
  int row = blockIdx.x;
  int tid = threadIdx.x;
  const float* dr = delta + (size_t)row * D_;
  float* hr = h + (size_t)row * D_;
  float x[3]; float s = 0.f;
#pragma unroll
  for (int t = 0; t < 3; ++t) { int d = tid + t * 256; x[t] = hr[d] + dr[d]; s += x[t]; }
  red[tid] = s; __syncthreads();
  for (int o = 128; o > 0; o >>= 1) { if (tid < o) red[tid] += red[tid + o]; __syncthreads(); }
  float mean = red[0] / (float)D_;
  __syncthreads();
  float vs = 0.f;
#pragma unroll
  for (int t = 0; t < 3; ++t) { float dv = x[t] - mean; vs += dv * dv; }
  red[tid] = vs; __syncthreads();
  for (int o = 128; o > 0; o >>= 1) { if (tid < o) red[tid] += red[tid + o]; __syncthreads(); }
  float rstd = rsqrtf(red[0] / (float)D_ + 1e-5f);
#pragma unroll
  for (int t = 0; t < 3; ++t) {
    int d = tid + t * 256;
    float y = (x[t] - mean) * rstd * g[d] + bvec[d];
    hr[d] = y;
    hb[(size_t)row * D_ + d] = (__bf16)y;
  }
}

// ---- classifier head: logits[b,c] = h[b,0,:] . cls_W[:,c] + cls_b[c] -------
__global__ void k_cls(const float* __restrict__ h, const float* __restrict__ W,
                      const float* __restrict__ bias, float* __restrict__ out)
{
  int t = threadIdx.x;
  if (t >= B_ * C_) return;
  int b = t / C_, c = t % C_;
  const float* hr = h + (size_t)(b * L_) * D_;   // sequence position 0 (CLS)
  float acc = bias[c];
  for (int d = 0; d < D_; ++d) acc += hr[d] * W[d * C_ + c];
  out[t] = acc;
}

// ===========================================================================
extern "C" void kernel_launch(void* const* d_in, const int* in_sizes, int n_in,
                              void* d_out, int out_size, void* d_ws, size_t ws_size,
                              hipStream_t stream)
{
  const int*           word_ids  = (const int*)d_in[0];
  const unsigned char* adj       = (const unsigned char*)d_in[1];  // np.bool_: 1 byte
  const int*           edge_types= (const int*)d_in[2];
  const float*         word_emb  = (const float*)d_in[3];
  const float*         edge_emb  = (const float*)d_in[4];
  const float *Wq = (const float*)d_in[5],  *bq = (const float*)d_in[6];
  const float *Wk = (const float*)d_in[7],  *bk = (const float*)d_in[8];
  const float *Wv = (const float*)d_in[9],  *bv = (const float*)d_in[10];
  const float *Wo = (const float*)d_in[11], *bo = (const float*)d_in[12];
  const float *ln1_g = (const float*)d_in[13], *ln1_b = (const float*)d_in[14];
  const float *W1 = (const float*)d_in[15], *b1 = (const float*)d_in[16];
  const float *W2 = (const float*)d_in[17], *b2 = (const float*)d_in[18];
  const float *ln2_g = (const float*)d_in[19], *ln2_b = (const float*)d_in[20];
  const float *cls_W = (const float*)d_in[21], *cls_b = (const float*)d_in[22];

  // ---- workspace carve-out (256B aligned) ----------------------------------
  char* ws = (char*)d_ws;
  size_t off = 0;
  auto alloc = [&](size_t bytes) -> char* {
    char* ptr = ws + off;
    off += (bytes + 255) & ~(size_t)255;
    return ptr;
  };
  const size_t WDD = (size_t)D_ * D_, WDF = (size_t)D_ * F_;
  const size_t layerW = 4 * WDD + 2 * WDF;
  __bf16* wts  = (__bf16*)alloc((size_t)NL_ * layerW * 2);          // 56.6 MB
  __bf16* epad = (__bf16*)alloc(64 * 64 * 2);
  float*  h    = (float*) alloc((size_t)B_ * L_ * D_ * 4);          // 12.6 MB
  __bf16* hb   = (__bf16*)alloc((size_t)B_ * L_ * D_ * 2);
  float*  tmp  = (float*) alloc((size_t)B_ * L_ * D_ * 4);          // reused
  __bf16* qb   = (__bf16*)alloc((size_t)B_ * H_ * L_ * HD_ * 2);
  __bf16* kb   = (__bf16*)alloc((size_t)B_ * H_ * L_ * HD_ * 2);
  __bf16* vtb  = (__bf16*)alloc((size_t)B_ * H_ * L_ * HD_ * 2);
  float*  pbuf = (float*) alloc((size_t)B_ * H_ * L_ * 64 * 4);     // edge proj
  float*  sc   = (float*) alloc((size_t)B_ * H_ * L_ * L_ * 4);     // 50 MB
  __bf16* attb = (__bf16*)alloc((size_t)B_ * H_ * L_ * L_ * 2);     // 25 MB
  __bf16* aob  = (__bf16*)alloc((size_t)B_ * L_ * D_ * 2);
  __bf16* ffab = (__bf16*)alloc((size_t)B_ * L_ * F_ * 2);          // 25 MB

  auto wq_t = [&](int l) { return wts + l * layerW; };
  auto wk_t = [&](int l) { return wts + l * layerW + WDD; };
  auto wv_t = [&](int l) { return wts + l * layerW + 2 * WDD; };
  auto wo_t = [&](int l) { return wts + l * layerW + 3 * WDD; };
  auto w1_t = [&](int l) { return wts + l * layerW + 4 * WDD; };
  auto w2_t = [&](int l) { return wts + l * layerW + 4 * WDD + WDF; };

  // batched direct GEMM (attention)
  auto gemm = [&](const __bf16* A_, const __bf16* Bt_, const float* bias_,
                  void* C_, int M, int N, int K, int ldc,
                  long long sA, long long sB, long long sC, int nb, int mode) {
    int tiles = (M / 32) * (N / 32);
    dim3 g((tiles + 7) / 8, nb);
    k_gemm<<<g, 256, 0, stream>>>(A_, Bt_, bias_, C_, M, N, K, ldc, sA, sB, sC, mode);
  };
  // LDS-tiled GEMM (batch=1, M%128==0, N%64==0, K%64==0)
  auto gemmL = [&](const __bf16* A_, const __bf16* Bt_, const float* bias_,
                   void* C_, int M, int N, int K, int ldc, int mode) {
    dim3 g(M / MB_, N / NB_);
    k_gemm_lds<<<g, 256, 0, stream>>>(A_, Bt_, bias_, C_, K, ldc, mode);
  };

  // ---- one-time (per launch) weight conversion/transposition --------------
  for (int l = 0; l < NL_; ++l) {
    const int gDD = (D_ / 32) * (D_ / 32);
    const int gDF = (D_ / 32) * (F_ / 32);
    k_wcvt<<<gDD, 256, 0, stream>>>(Wq + (size_t)l * WDD, wq_t(l), D_, D_);
    k_wcvt<<<gDD, 256, 0, stream>>>(Wk + (size_t)l * WDD, wk_t(l), D_, D_);
    k_wcvt<<<gDD, 256, 0, stream>>>(Wv + (size_t)l * WDD, wv_t(l), D_, D_);
    k_wcvt<<<gDD, 256, 0, stream>>>(Wo + (size_t)l * WDD, wo_t(l), D_, D_);
    k_wcvt<<<gDF, 256, 0, stream>>>(W1 + (size_t)l * WDF, w1_t(l), D_, F_);
    k_wcvt<<<gDF, 256, 0, stream>>>(W2 + (size_t)l * WDF, w2_t(l), F_, D_);
  }
  k_edgepad<<<16, 256, 0, stream>>>(edge_emb, epad);

  // ---- embeddings ----------------------------------------------------------
  k_embed<<<B_ * L_, 256, 0, stream>>>(word_ids, word_emb, h, hb);

  const int ML = B_ * L_;                        // 4096 rows
  const long long QS = (long long)L_ * HD_;      // per-(b,h) q/k/v stride
  const int packBlocks = (ML * D_) / 256;

  for (int l = 0; l < NL_; ++l) {
    // q, k, v projections (M=4096, N=768, K=768) + head repack
    gemmL(hb, wq_t(l), bq + l * D_, tmp, ML, D_, D_, D_, 0);
    k_pack_heads<<<packBlocks, 256, 0, stream>>>(tmp, qb);
    gemmL(hb, wk_t(l), bk + l * D_, tmp, ML, D_, D_, D_, 0);
    k_pack_heads<<<packBlocks, 256, 0, stream>>>(tmp, kb);
    gemmL(hb, wv_t(l), bv + l * D_, tmp, ML, D_, D_, D_, 0);
    k_pack_vT<<<packBlocks, 256, 0, stream>>>(tmp, vtb);

    // scores = q @ k^T   (batch = B*H, M=N=256, K=64)
    gemm(qb, kb, nullptr, sc, L_, L_, HD_, L_, QS, QS, (long long)L_ * L_, B_ * H_, 0);
    // edge projections p[b,h,i,e] = q . edge_emb[e]   (N padded 50->64)
    gemm(qb, epad, nullptr, pbuf, L_, 64, HD_, 64, QS, 0, (long long)L_ * 64, B_ * H_, 0);

    // fused edge-gather + mask + softmax -> bf16 attention
    k_softmax<<<B_ * H_ * L_, 256, 0, stream>>>(sc, pbuf, edge_types, adj, attb);

    // out = att @ v   (batch = B*H, M=256, N=64, K=256) -> tmp as [B,H,L,HD]
    gemm(attb, vtb, nullptr, tmp, L_, HD_, L_, HD_, (long long)L_ * L_, QS, QS, B_ * H_, 0);
    k_pack_merge<<<packBlocks, 256, 0, stream>>>(tmp, aob);

    // output projection + residual + LN1
    gemmL(aob, wo_t(l), bo + l * D_, tmp, ML, D_, D_, D_, 0);
    k_ln<<<ML, 256, 0, stream>>>(tmp, h, hb, ln1_g + l * D_, ln1_b + l * D_);

    // FFN: relu(h@W1+b1)@W2+b2, ReLU fused into GEMM1 epilogue (bf16 out)
    gemmL(hb, w1_t(l), b1 + l * F_, ffab, ML, F_, D_, F_, 2);
    gemmL(ffab, w2_t(l), b2 + l * D_, tmp, ML, D_, F_, D_, 0);
    k_ln<<<ML, 256, 0, stream>>>(tmp, h, hb, ln2_g + l * D_, ln2_b + l * D_);
  }

  // classifier head on CLS token (32 outputs)
  k_cls<<<1, 32, 0, stream>>>(h, cls_W, cls_b, (float*)d_out);
}
